// MambaMixer_1683627180238
// MI455X (gfx1250) — matrix operand order repaired and checked by reading, not compiled, not run on hardware
//
#include <hip/hip_runtime.h>
#include <hip/hip_bf16.h>

typedef __attribute__((ext_vector_type(16))) __bf16 v16bf;
typedef __attribute__((ext_vector_type(8)))  __bf16 v8bf;
typedef __attribute__((ext_vector_type(8)))  float  v8f;
typedef __attribute__((ext_vector_type(4)))  unsigned int v4ui;
typedef __attribute__((ext_vector_type(8)))  int v8i;
typedef __attribute__((ext_vector_type(4)))  int v4i;

#define B_SZ 8
#define T_SZ 2048
#define D_SZ 1024
#define L_SZ 2

#define WMMA_BF16(a, b, c)                                                   \
  __builtin_amdgcn_wmma_f32_16x16x32_bf16(false, (a), false, (b), (short)0,  \
                                          (c), false, false)

// ---------------- WMMA fragment loaders (contiguous k order, shared by A and B) ----
// A matrix 16x32 bf16: lane row = lane&15; lanes 0-15 hold K 0..15,
// lanes 16-31 hold K 16..31  -> two 16-byte contiguous chunks.
__device__ __forceinline__ v16bf load_a_frag(const __bf16* __restrict__ A,
                                             int lda, int mbase, int kbase,
                                             int lane) {
  const int row = lane & 15;
  const int k0  = (lane >> 4) * 16;
  const __bf16* p = A + (size_t)(mbase + row) * lda + kbase + k0;
  v8bf lo = *(const v8bf*)(p);
  v8bf hi = *(const v8bf*)(p + 8);
  return __builtin_shufflevector(lo, hi, 0, 1, 2, 3, 4, 5, 6, 7,
                                 8, 9, 10, 11, 12, 13, 14, 15);
}

// B matrix 32x16 bf16: lane col = lane&15; lanes 0-15 hold K 0..15,
// lanes 16-31 hold K 16..31 -> one contiguous 32-byte chunk of W row `col`.
__device__ __forceinline__ v16bf load_b_frag(const __bf16* __restrict__ W,
                                             int ldw, int nbase, int kbase,
                                             int lane) {
  const int col = lane & 15;
  const int k0  = (lane >> 4) * 16;
  return *(const v16bf*)(W + (size_t)(nbase + col) * ldw + kbase + k0);
}

// 16x64 register-blocked WMMA main loop over K = D_SZ, 2x-unrolled with two
// static buffer sets (no inter-buffer register copies) and a peeled tail.
// Loads for one buffer are issued while WMMAs consume the other.
__device__ __forceinline__ void mainloop_16x64(const __bf16* __restrict__ A,
                                               const __bf16* __restrict__ W,
                                               int lda, int ldw, int mtile,
                                               int ntile, int lane,
                                               v8f acc[4]) {
  v16bf a0 = load_a_frag(A, lda, mtile, 0, lane);
  v16bf b0[4];
#pragma unroll
  for (int j = 0; j < 4; ++j)
    b0[j] = load_b_frag(W, ldw, ntile + 16 * j, 0, lane);
#pragma unroll 1
  for (int k = 0; k < D_SZ - 64; k += 64) {
    v16bf a1 = load_a_frag(A, lda, mtile, k + 32, lane);
    v16bf b1[4];
#pragma unroll
    for (int j = 0; j < 4; ++j)
      b1[j] = load_b_frag(W, ldw, ntile + 16 * j, k + 32, lane);
#pragma unroll
    for (int j = 0; j < 4; ++j) acc[j] = WMMA_BF16(a0, b0[j], acc[j]);
    a0 = load_a_frag(A, lda, mtile, k + 64, lane);
#pragma unroll
    for (int j = 0; j < 4; ++j)
      b0[j] = load_b_frag(W, ldw, ntile + 16 * j, k + 64, lane);
#pragma unroll
    for (int j = 0; j < 4; ++j) acc[j] = WMMA_BF16(a1, b1[j], acc[j]);
  }
  {  // tail: a0/b0 hold K-block D_SZ-64; load and do D_SZ-32
    v16bf a1 = load_a_frag(A, lda, mtile, D_SZ - 32, lane);
    v16bf b1[4];
#pragma unroll
    for (int j = 0; j < 4; ++j)
      b1[j] = load_b_frag(W, ldw, ntile + 16 * j, D_SZ - 32, lane);
#pragma unroll
    for (int j = 0; j < 4; ++j) acc[j] = WMMA_BF16(a0, b0[j], acc[j]);
#pragma unroll
    for (int j = 0; j < 4; ++j) acc[j] = WMMA_BF16(a1, b1[j], acc[j]);
  }
}

// ---------------- weight fp32 -> bf16 cast -------------------------------
__global__ void cast_f32_bf16(const float* __restrict__ src,
                              __bf16* __restrict__ dst, int n) {
  int i = blockIdx.x * blockDim.x + threadIdx.x;
  if (i < n) dst[i] = (__bf16)src[i];
}

// ---------------- LayerNorm -> bf16 --------------------------------------
__global__ void __launch_bounds__(256)
layernorm_bf16(const float* __restrict__ X, const float* __restrict__ g,
               const float* __restrict__ bb, __bf16* __restrict__ Y) {
  __shared__ float s_sum[256];
  __shared__ float s_sq[256];
  const int row  = blockIdx.x;
  const int base = threadIdx.x * 4;
  const float* x = X + (size_t)row * D_SZ;
  float v[4], s = 0.f, sq = 0.f;
#pragma unroll
  for (int i = 0; i < 4; ++i) {
    v[i] = x[base + i];
    s += v[i];
    sq += v[i] * v[i];
  }
  s_sum[threadIdx.x] = s;
  s_sq[threadIdx.x]  = sq;
  __syncthreads();
  for (int off = 128; off > 0; off >>= 1) {
    if ((int)threadIdx.x < off) {
      s_sum[threadIdx.x] += s_sum[threadIdx.x + off];
      s_sq[threadIdx.x]  += s_sq[threadIdx.x + off];
    }
    __syncthreads();
  }
  const float mean = s_sum[0] * (1.0f / D_SZ);
  const float var  = s_sq[0] * (1.0f / D_SZ) - mean * mean;
  const float inv  = rsqrtf(var + 1e-5f);
  __bf16* y = Y + (size_t)row * D_SZ;
#pragma unroll
  for (int i = 0; i < 4; ++i) {
    const int c = base + i;
    y[c] = (__bf16)((v[i] - mean) * inv * g[c] + bb[c]);
  }
}

// ---------------- big parallel GEMM: C[M,N] = A[M,K] @ W[N,K]^T + bias ---
// Block = 4 waves, each wave owns a 16x64 tile -> block tile 16x256.
__global__ void __launch_bounds__(128)
gemm_inproj(const __bf16* __restrict__ A, const __bf16* __restrict__ W,
            const float* __restrict__ bias, float* __restrict__ C, int N) {
  const int lane  = threadIdx.x & 31;
  const int wave  = threadIdx.x >> 5;
  const int ntile = blockIdx.x * 256 + wave * 64;
  const int mtile = blockIdx.y * 16;
  v8f acc[4] = {};
  mainloop_16x64(A, W, D_SZ, D_SZ, mtile, ntile, lane, acc);
  const int rbase = (lane >> 4) * 8;
#pragma unroll
  for (int j = 0; j < 4; ++j) {
    const int col  = ntile + 16 * j + (lane & 15);
    const float bv = bias[col];
#pragma unroll
    for (int r = 0; r < 8; ++r) {
      const int row = mtile + rbase + r;
      C[(size_t)row * N + col] = acc[j][r] + bv;
    }
  }
}

// ---------------- recurrence init ----------------------------------------
__global__ void init_recurrence(__bf16* __restrict__ h0,
                                unsigned int* __restrict__ bar) {
  int i = blockIdx.x * blockDim.x + threadIdx.x;
  if (i < 16 * D_SZ) h0[i] = (__bf16)0.0f;
  if (i == 0) *bar = 0u;
}

// ---------------- serial GRU-style recurrence (persistent, 16 WGs) -------
// WG w owns output columns [64w, 64w+64); its W_h slice (64x1024 bf16 =
// 128 KB) is staged once into LDS via the Tensor Data Mover and stays
// resident for all 2048 steps.
__global__ void __launch_bounds__(128)
recurrence(const __bf16* __restrict__ Wh, const float* __restrict__ bh,
           const float* __restrict__ uv, __bf16* __restrict__ hall,
           unsigned int* __restrict__ bar) {
  extern __shared__ __align__(16) unsigned char smem_raw[];
  __bf16* sW = (__bf16*)smem_raw;  // [64][1024]

  const int lane    = threadIdx.x & 31;
  const int wave    = threadIdx.x >> 5;
  const int wg      = blockIdx.x;  // 0..15
  const int nbase_g = wg * 64;

#if __has_builtin(__builtin_amdgcn_tensor_load_to_lds)
  if (wave == 0) {  // one wave issues the TDM descriptor (EXEC ignored)
    const unsigned long long ga =
        (unsigned long long)(Wh + (size_t)nbase_g * D_SZ);
    // D# group0: count=1 | lds_addr=0 | global_addr | type=2 ("image")
    v4ui g0 = {1u, 0u, (unsigned)(ga & 0xffffffffu),
               (unsigned)((ga >> 32) & 0x01ffffffu) | (2u << 30)};
    // D# group1: data_size=2B; tensor_dim0=1024, tensor_dim1=1024;
    // tile_dim0=1024, tile_dim1=64; tensor_dim0_stride=1024.
    v8i g1 = {(int)(1u << 16),     // data_size code 1 (2 bytes)
              (int)(1024u << 16),  // tensor_dim0[15:0]
              (int)(1024u << 16),  // tensor_dim0 hi | tensor_dim1[15:0]
              (int)(1024u << 16),  // tensor_dim1 hi | tile_dim0=1024
              64,                  // tile_dim1=64 | tile_dim2=0
              1024, 0, 0};         // tensor_dim0_stride=1024
    v4i z = {0, 0, 0, 0};
#if __clang_major__ >= 23
    v8i z8 = {0, 0, 0, 0, 0, 0, 0, 0};
    __builtin_amdgcn_tensor_load_to_lds(g0, g1, z, z, z8, 0);
#else
    __builtin_amdgcn_tensor_load_to_lds(g0, g1, z, z, 0);
#endif
    __builtin_amdgcn_s_wait_tensorcnt(0);
  }
  __syncthreads();
#else
  {  // fallback: stage W_h slice into LDS with 16B vector copies
    const uint4* src = (const uint4*)(Wh + (size_t)nbase_g * D_SZ);
    uint4* dst = (uint4*)sW;
    const int total = 64 * D_SZ * 2 / 16;
    for (int i = threadIdx.x; i < total; i += blockDim.x) dst[i] = src[i];
  }
  __syncthreads();
#endif

  const int col   = lane & 15;
  const int rbase = (lane >> 4) * 8;
  const int nloc  = wave * 16 + col;  // column within this WG's 64-slice
  const int ng    = nbase_g + nloc;   // global column
  const float bhv = bh[ng];
  const __bf16* sB = sW + (size_t)nloc * D_SZ + ((lane >> 4) * 16);

  for (int t = 0; t < T_SZ; ++t) {
    const __bf16* hcur = hall + (size_t)t * (16 * D_SZ);
    v8f acc = {};
    v16bf a0 = load_a_frag(hcur, D_SZ, 0, 0, lane);
#pragma unroll 1
    for (int k = 0; k < D_SZ - 64; k += 64) {
      v16bf a1 = load_a_frag(hcur, D_SZ, 0, k + 32, lane);
      acc = WMMA_BF16(a0, *(const v16bf*)(sB + k), acc);
      a0 = load_a_frag(hcur, D_SZ, 0, k + 64, lane);
      acc = WMMA_BF16(a1, *(const v16bf*)(sB + k + 32), acc);
    }
    {
      v16bf a1 = load_a_frag(hcur, D_SZ, 0, D_SZ - 32, lane);
      acc = WMMA_BF16(a0, *(const v16bf*)(sB + D_SZ - 64), acc);
      acc = WMMA_BF16(a1, *(const v16bf*)(sB + D_SZ - 32), acc);
    }
    __bf16* hnext = hall + (size_t)(t + 1) * (16 * D_SZ);
#pragma unroll
    for (int r = 0; r < 8; ++r) {
      const int m = rbase + r;  // padded batch row 0..15; valid < 8
      float hn = 0.0f;
      if (m < B_SZ) {
        const size_t rowuv = ((size_t)m * T_SZ + t) * (2 * D_SZ);
        const float u     = uv[rowuv + ng];
        const float vv    = uv[rowuv + D_SZ + ng];
        const float hprev = (float)hcur[(size_t)m * D_SZ + ng];
        const float gate  = 1.0f / (1.0f + __expf(-u));
        const float cand  = tanhf(vv + acc[r] + bhv);
        hn = gate * cand + (1.0f - gate) * hprev;
      }
      hnext[(size_t)m * D_SZ + ng] = (__bf16)hn;
    }
    // grid-wide barrier: monotonic arrival counter, 16 resident WGs
    __syncthreads();
    if (threadIdx.x == 0) {
      __threadfence();
      atomicAdd(bar, 1u);
      const unsigned int target = 16u * (unsigned int)(t + 1);
      while (__hip_atomic_load(bar, __ATOMIC_ACQUIRE,
                               __HIP_MEMORY_SCOPE_AGENT) < target) {
        __builtin_amdgcn_s_sleep(1);
      }
    }
    __syncthreads();
  }
}

// ---------------- out-projection + residual (parallel over all t) --------
__global__ void __launch_bounds__(128)
outproj(const __bf16* __restrict__ hall, const __bf16* __restrict__ Wout,
        const float* __restrict__ bout, const float* __restrict__ resid,
        float* __restrict__ out) {
  const int lane  = threadIdx.x & 31;
  const int wave  = threadIdx.x >> 5;
  const int t     = blockIdx.y;
  const int ntile = blockIdx.x * 256 + wave * 64;
  const __bf16* A = hall + (size_t)(t + 1) * (16 * D_SZ);
  v8f acc[4] = {};
  mainloop_16x64(A, Wout, D_SZ, D_SZ, 0, ntile, lane, acc);
  const int rbase = (lane >> 4) * 8;
#pragma unroll
  for (int j = 0; j < 4; ++j) {
    const int col  = ntile + 16 * j + (lane & 15);
    const float bv = bout[col];
#pragma unroll
    for (int r = 0; r < 8; ++r) {
      const int m = rbase + r;
      if (m < B_SZ) {
        const size_t idx = ((size_t)m * T_SZ + t) * D_SZ + col;
        out[idx] = acc[j][r] + bv + resid[idx];
      }
    }
  }
}

// ---------------- host-side orchestration --------------------------------
extern "C" void kernel_launch(void* const* d_in, const int* in_sizes, int n_in,
                              void* d_out, int out_size, void* d_ws,
                              size_t ws_size, hipStream_t stream) {
  const float* x     = (const float*)d_in[0];
  const float* ln_g  = (const float*)d_in[1];
  const float* ln_b  = (const float*)d_in[2];
  const float* W_in  = (const float*)d_in[3];
  const float* b_in  = (const float*)d_in[4];
  const float* W_h   = (const float*)d_in[5];
  const float* b_h   = (const float*)d_in[6];
  const float* W_out = (const float*)d_in[7];
  const float* b_out = (const float*)d_in[8];
  float* out = (float*)d_out;

  unsigned char* ws = (unsigned char*)d_ws;
  size_t off = 0;
  auto alloc = [&](size_t bytes) -> void* {
    off = (off + 255) & ~(size_t)255;
    void* p = ws + off;
    off += bytes;
    return p;
  };
  __bf16* xn    = (__bf16*)alloc((size_t)B_SZ * T_SZ * D_SZ * 2);
  float*  uv    = (float*)alloc((size_t)B_SZ * T_SZ * 2 * D_SZ * 4);
  __bf16* hall  = (__bf16*)alloc((size_t)(T_SZ + 1) * 16 * D_SZ * 2);
  float*  act   = (float*)alloc((size_t)B_SZ * T_SZ * D_SZ * 4);
  __bf16* wInB  = (__bf16*)alloc((size_t)2 * D_SZ * D_SZ * 2);
  __bf16* wHB   = (__bf16*)alloc((size_t)D_SZ * D_SZ * 2);
  __bf16* wOB   = (__bf16*)alloc((size_t)D_SZ * D_SZ * 2);
  unsigned int* bar = (unsigned int*)alloc(256);
  (void)ws_size; (void)in_sizes; (void)n_in; (void)out_size;

  const int ROWS = B_SZ * T_SZ;  // 16384

  for (int l = 0; l < L_SZ; ++l) {
    const float* in_act  = (l == 0) ? x : act;
    float*       out_act = (l == L_SZ - 1) ? out : act;

    const int nIn = 2 * D_SZ * D_SZ;
    const int nSq = D_SZ * D_SZ;
    cast_f32_bf16<<<dim3((nIn + 255) / 256), 256, 0, stream>>>(
        W_in + (size_t)l * nIn, wInB, nIn);
    cast_f32_bf16<<<dim3((nSq + 255) / 256), 256, 0, stream>>>(
        W_h + (size_t)l * nSq, wHB, nSq);
    cast_f32_bf16<<<dim3((nSq + 255) / 256), 256, 0, stream>>>(
        W_out + (size_t)l * nSq, wOB, nSq);

    layernorm_bf16<<<dim3(ROWS), 256, 0, stream>>>(
        in_act, ln_g + (size_t)l * D_SZ, ln_b + (size_t)l * D_SZ, xn);

    gemm_inproj<<<dim3(2 * D_SZ / 256, ROWS / 16), 128, 0, stream>>>(
        xn, wInB, b_in + (size_t)l * 2 * D_SZ, uv, 2 * D_SZ);

    init_recurrence<<<dim3((16 * D_SZ + 255) / 256), 256, 0, stream>>>(hall,
                                                                       bar);

    recurrence<<<dim3(16), 128, 64 * D_SZ * 2, stream>>>(
        wHB, b_h + (size_t)l * D_SZ, uv, hall, bar);

    outproj<<<dim3(D_SZ / 256, T_SZ), 128, 0, stream>>>(
        hall, wOB, b_out + (size_t)l * D_SZ, in_act, out_act);
  }
}
